// Parallax_attention_module_85255100826212
// MI455X (gfx1250) — matrix operand order repaired and checked by reading, not compiled, not run on hardware
//
#include <hip/hip_runtime.h>

// ---------------------------------------------------------------------------
// Parallax attention module for MI455X (gfx1250, wave32, WMMA).
// Fused per-(direction,row) kernel: BN-folded 1x1-conv projections and the
// 256x256x64 row-attention GEMM both run on v_wmma_f32_16x16x32_f16; softmax
// is LDS-resident; the 256MB attention output is streamed to HBM exactly once
// (memory-roofline floor ~12us at 23.3 TB/s). Morphology = 2 tiny kernels.
// ---------------------------------------------------------------------------

typedef _Float16 v16h __attribute__((ext_vector_type(16)));
typedef _Float16 v8h  __attribute__((ext_vector_type(8)));
typedef float    v8f  __attribute__((ext_vector_type(8)));

#define WMMA_F16(a, b, c) \
  __builtin_amdgcn_wmma_f32_16x16x32_f16(false, (a), false, (b), (short)0, (c), false, false)

#define XS 72    // row stride (halfs) for X/Q/S and weight tiles (16B-aligned rows)
#define SS 264   // score row stride (floats)

// A-matrix 16x32 f16 fragment (M = row per lane 0..15; hi half-wave selects
// K in {hi*8..hi*8+7, 16+hi*8..23+hi*8}) -> two aligned b128 LDS loads.
static __device__ __forceinline__ v16h frag_a(const _Float16* row, int hi, int kb) {
  const v8h* p0 = (const v8h*)(row + kb + hi * 8);
  const v8h* p1 = (const v8h*)(row + kb + 16 + hi * 8);
  v8h lo = *p0, hh = *p1;
  v16h r;
#pragma unroll
  for (int e = 0; e < 8; ++e) { r[e] = lo[e]; r[e + 8] = hh[e]; }
  return r;
}

// B-matrix 32x16 f16 fragment (N = col per lane 0..15; hi half-wave holds
// K in {hi*8..hi*8+7, 16+hi*8..23+hi*8}) -> two aligned b128 LDS loads.
static __device__ __forceinline__ v16h frag_b(const _Float16* row, int hi, int kb) {
  const v8h* p0 = (const v8h*)(row + kb + hi * 8);
  const v8h* p1 = (const v8h*)(row + kb + 16 + hi * 8);
  v8h lo = *p0, hh = *p1;
  v16h r;
#pragma unroll
  for (int e = 0; e < 8; ++e) { r[e] = lo[e]; r[e + 8] = hh[e]; }
  return r;
}

__global__ __launch_bounds__(256)
void pam_attn(const float* __restrict__ xl, const float* __restrict__ xr,
              const float* __restrict__ Wq, const float* __restrict__ qg,
              const float* __restrict__ qb, const float* __restrict__ qm,
              const float* __restrict__ qv, const float* __restrict__ Wk,
              const float* __restrict__ kg, const float* __restrict__ kb_,
              const float* __restrict__ km, const float* __restrict__ kv,
              float* __restrict__ out, float* __restrict__ occ) {
  // LDS map (139 KB total; score strip overlays the dead weight region):
  //   [0      .. 36864)  Xq rows -> reused in-place as Q[w][c] (f16, stride XS)
  //   [36864  .. 73728)  Xk rows -> reused in-place as S[v][c]
  //   [73728  .. 82944)  folded Wq' (f16)    } dead after projection;
  //   [82944  .. 92160)  folded Wk' (f16)    } overlaid by score strip
  //   [92160  .. 92672)  biases (f32)        }
  //   [73728  .. 141312) score strip 64x264 f32
  //   [141312 .. 142336) softmax reduction scratch
  __shared__ __align__(16) char smem[142336];
  _Float16* sQ  = (_Float16*)(smem);
  _Float16* sS  = (_Float16*)(smem + 36864);
  _Float16* sWq = (_Float16*)(smem + 73728);
  _Float16* sWk = (_Float16*)(smem + 82944);
  float* sBq    = (float*)(smem + 92160);
  float* sBk    = (float*)(smem + 92416);
  float* sScore = (float*)(smem + 73728);
  float* sRed   = (float*)(smem + 141312);

  const int tid  = threadIdx.x;
  const int dir  = blockIdx.x >> 9;   // 0: M_right_to_left, 1: M_left_to_right
  const int n    = blockIdx.x & 511;  // b*H + h
  const int b    = n >> 7;
  const int h    = n & 127;
  const int wv   = tid >> 5;
  const int lane = tid & 31;
  const int lm   = lane & 15;
  const int hi   = lane >> 4;

  // ---- Fold BN into weights: W'[o,c] = s_o*W[o,c], bias_o = beta - mean*s_o.
  for (int i = tid; i < 4096; i += 256) {
    int o = i >> 6, c = i & 63;
    float sq = qg[o] * rsqrtf(qv[o] + 1e-5f);
    float sk = kg[o] * rsqrtf(kv[o] + 1e-5f);
    sWq[o * XS + c] = (_Float16)(Wq[i] * sq);
    sWk[o * XS + c] = (_Float16)(Wk[i] * sk);
  }
  if (tid < 64) {
    float sq = qg[tid] * rsqrtf(qv[tid] + 1e-5f);
    float sk = kg[tid] * rsqrtf(kv[tid] + 1e-5f);
    sBq[tid] = qb[tid] - qm[tid] * sq;
    sBk[tid] = kb_[tid] - km[tid] * sk;
  }

  // ---- Stage this image row of both inputs as f16 [w][c] (coalesced reads).
  const float* xq = dir ? xr : xl;
  const float* xk = dir ? xl : xr;
  for (int c = 0; c < 64; ++c) {
    size_t g = ((size_t)(b * 64 + c) * 128 + h) * 256 + tid;
    sQ[tid * XS + c] = (_Float16)xq[g];
    sS[tid * XS + c] = (_Float16)xk[g];
  }
  __syncthreads();

  // ---- Projections via WMMA. Wave wv owns spatial rows [32wv, 32wv+32), so
  // it can overwrite its X rows with Q/S in place (per-wave LDS ops stay in
  // program order; all 4 output column-tiles are accumulated before stores).
#pragma unroll 1
  for (int pass = 0; pass < 2; ++pass) {
    _Float16* sX        = pass ? sS : sQ;
    const _Float16* sW  = pass ? sWk : sWq;
    const float* sB     = pass ? sBk : sBq;
#pragma unroll 1
    for (int half_ = 0; half_ < 2; ++half_) {
      int mt = 2 * wv + half_;
      const _Float16* arow = sX + (mt * 16 + lm) * XS;
      v8f a0 = {}, a1 = {}, a2 = {}, a3 = {};
#pragma unroll
      for (int kb2 = 0; kb2 < 64; kb2 += 32) {
        v16h a  = frag_a(arow, hi, kb2);
        v16h b0 = frag_b(sW + (0 * 16 + lm) * XS, hi, kb2);
        a0 = WMMA_F16(a, b0, a0);
        v16h b1 = frag_b(sW + (1 * 16 + lm) * XS, hi, kb2);
        a1 = WMMA_F16(a, b1, a1);
        v16h b2 = frag_b(sW + (2 * 16 + lm) * XS, hi, kb2);
        a2 = WMMA_F16(a, b2, a2);
        v16h b3 = frag_b(sW + (3 * 16 + lm) * XS, hi, kb2);
        a3 = WMMA_F16(a, b3, a3);
      }
      _Float16* drow = sX + (mt * 16 + hi * 8) * XS;
      float bb0 = sB[lm], bb1 = sB[16 + lm], bb2 = sB[32 + lm], bb3 = sB[48 + lm];
#pragma unroll
      for (int r = 0; r < 8; ++r) {
        drow[r * XS + lm]      = (_Float16)(a0[r] + bb0);
        drow[r * XS + 16 + lm] = (_Float16)(a1[r] + bb1);
        drow[r * XS + 32 + lm] = (_Float16)(a2[r] + bb2);
        drow[r * XS + 48 + lm] = (_Float16)(a3[r] + bb3);
      }
    }
  }
  __syncthreads();

  // ---- Attention: 4 strips of 64 query rows; WMMA scores -> LDS softmax ->
  // single coalesced HBM store; thread t owns output column t => colsum free.
  float cs = 0.f;
#pragma unroll 1
  for (int strip = 0; strip < 4; ++strip) {
#pragma unroll 1
    for (int j = 0; j < 8; ++j) {
      int t   = wv * 8 + j;     // 64 tiles of the 64x256 strip
      int mtl = t >> 4;         // 0..3
      int vt  = t & 15;         // 0..15
      const _Float16* arow = sQ + (strip * 64 + mtl * 16 + lm) * XS;
      const _Float16* brow = sS + (vt * 16 + lm) * XS;
      v8f acc = {};
#pragma unroll
      for (int kb2 = 0; kb2 < 64; kb2 += 32) {
        v16h a  = frag_a(arow, hi, kb2);
        v16h bv = frag_b(brow, hi, kb2);
        acc = WMMA_F16(a, bv, acc);
      }
      float* sp = sScore + (mtl * 16 + hi * 8) * SS + vt * 16 + lm;
#pragma unroll
      for (int r = 0; r < 8; ++r) sp[r * SS] = acc[r];
    }
    __syncthreads();

    // softmax over each of the 64 rows (4 threads per row)
    {
      int r = tid >> 2, part = tid & 3;
      float* rowp = sScore + r * SS + part * 64;
      float mx = -3.0e38f;
#pragma unroll 4
      for (int i = 0; i < 64; ++i) mx = fmaxf(mx, rowp[i]);
      sRed[r * 4 + part] = mx;
      __syncthreads();
      float m4 = fmaxf(fmaxf(sRed[r * 4], sRed[r * 4 + 1]),
                       fmaxf(sRed[r * 4 + 2], sRed[r * 4 + 3]));
      float sum = 0.f;
#pragma unroll 4
      for (int i = 0; i < 64; ++i) {
        float e = __expf(rowp[i] - m4);
        rowp[i] = e;
        sum += e;
      }
      __syncthreads();
      sRed[r * 4 + part] = sum;
      __syncthreads();
      float tot = sRed[r * 4] + sRed[r * 4 + 1] + sRed[r * 4 + 2] + sRed[r * 4 + 3];
      float inv = 1.f / tot;
#pragma unroll 4
      for (int i = 0; i < 64; ++i) rowp[i] *= inv;
    }
    __syncthreads();

    float* op = out + (size_t)dir * 33554432 + (size_t)n * 65536 + (size_t)strip * 16384;
#pragma unroll 4
    for (int k = 0; k < 64; ++k) {
      float vv = sScore[k * SS + tid];
      op[k * 256 + tid] = vv;
      cs += vv;  // column-sum over all 256 query rows for column `tid`
    }
    __syncthreads();
  }

  // occlusion = !(colsum > 0.1)
  occ[(size_t)dir * 131072 + (size_t)n * 256 + tid] = (cs > 0.1f) ? 0.f : 1.f;
}

// ---- morphology: binary closing with disk(3) on the inverted valid mask ----
__global__ __launch_bounds__(256)
void pam_dilate(const float* __restrict__ occ, float* __restrict__ dil) {
  int p   = blockIdx.x * 256 + threadIdx.x;  // 0..262143  (2 x B x H x W)
  int d   = p >> 17;
  int rem = p & 131071;
  int b   = rem >> 15;
  int yx  = rem & 32767;
  int y   = yx >> 8, x = yx & 255;
  const float* base = occ + (size_t)d * 131072 + (size_t)b * 32768;
  float any = 0.f;
#pragma unroll
  for (int dy = -3; dy <= 3; ++dy) {
#pragma unroll
    for (int dx = -3; dx <= 3; ++dx) {
      if (dy * dy + dx * dx <= 9) {
        int yy = y + dy, xx = x + dx;
        if (yy >= 0 && yy < 128 && xx >= 0 && xx < 256)
          if (base[yy * 256 + xx] > 0.5f) any = 1.f;
      }
    }
  }
  dil[p] = any;
}

__global__ __launch_bounds__(256)
void pam_erode(const float* __restrict__ dil, float* __restrict__ out) {
  int p   = blockIdx.x * 256 + threadIdx.x;
  int d   = p >> 17;
  int rem = p & 131071;
  int b   = rem >> 15;
  int yx  = rem & 32767;
  int y   = yx >> 8, x = yx & 255;
  const float* base = dil + (size_t)d * 131072 + (size_t)b * 32768;
  int cnt = 0;
#pragma unroll
  for (int dy = -3; dy <= 3; ++dy) {
#pragma unroll
    for (int dx = -3; dx <= 3; ++dx) {
      if (dy * dy + dx * dx <= 9) {
        int yy = y + dy, xx = x + dx;
        if (yy >= 0 && yy < 128 && xx >= 0 && xx < 256)
          if (base[yy * 256 + xx] > 0.5f) cnt++;
      }
    }
  }
  // zero padding => erosion passes only if all 29 in-bounds taps are set
  float V = (cnt == 29) ? 0.f : 1.f;  // V = 1 - closed
  // dir 1 -> V_l2r (first), dir 0 -> V_r2l (second)
  float* dst = out + 67108864 + (size_t)(d == 1 ? 0 : 131072) +
               ((size_t)b * 32768 + (size_t)y * 256 + x);
  *dst = V;
}

extern "C" void kernel_launch(void* const* d_in, const int* in_sizes, int n_in,
                              void* d_out, int out_size, void* d_ws, size_t ws_size,
                              hipStream_t stream) {
  (void)in_sizes; (void)n_in; (void)out_size; (void)ws_size;
  const float* xl = (const float*)d_in[0];
  const float* xr = (const float*)d_in[1];
  const float* Wq = (const float*)d_in[2];
  const float* qg = (const float*)d_in[3];
  const float* qb = (const float*)d_in[4];
  const float* qm = (const float*)d_in[5];
  const float* qv = (const float*)d_in[6];
  const float* Wk = (const float*)d_in[7];
  const float* kg = (const float*)d_in[8];
  const float* kb = (const float*)d_in[9];
  const float* km = (const float*)d_in[10];
  const float* kv = (const float*)d_in[11];
  float* out = (float*)d_out;
  float* occ = (float*)d_ws;          // 2 x 131072 floats
  float* dil = occ + 262144;          // 2 x 131072 floats

  pam_attn<<<dim3(1024), dim3(256), 0, stream>>>(xl, xr, Wq, qg, qb, qm, qv,
                                                 Wk, kg, kb, km, kv, out, occ);
  pam_dilate<<<dim3(1024), dim3(256), 0, stream>>>(occ, dil);
  pam_erode<<<dim3(1024), dim3(256), 0, stream>>>(dil, out);
}
